// model19_14611478741161
// MI455X (gfx1250) — matrix operand order repaired and checked by reading, not compiled, not run on hardware
//
#include <hip/hip_runtime.h>
#include <math.h>

typedef __attribute__((ext_vector_type(16))) _Float16 v16h;
typedef __attribute__((ext_vector_type(8)))  float    v8f;

#define N_NODES 19
#define KPAD    32
#define FC_OUT  1600
#define NTILES  (FC_OUT / 16)          // 100
#define WT_OFFSET    2048              // byte offset of Wt inside workspace
#define AFRAG_OFFSET 104448            // 2048 + 102400, multiple of 256

static __device__ __forceinline__ float fast_tanh(float x) {
#if __has_builtin(__builtin_amdgcn_tanhf)
    return __builtin_amdgcn_tanhf(x);   // hardware v_tanh_f32 (gfx1250 TRANS op)
#else
    return tanhf(x);
#endif
}

// ---------------------------------------------------------------------------
// Prep 1: build the 19x19 propagation matrix P = D^{-1/2} (A+I) D^{-1/2}
// ---------------------------------------------------------------------------
__global__ void build_P_kernel(const long long* __restrict__ ei, int E,
                               float* __restrict__ P) {
    if (blockIdx.x != 0 || threadIdx.x != 0) return;
    float deg[N_NODES];
    for (int i = 0; i < N_NODES; ++i) deg[i] = 1.0f;          // self loops
    for (int e = 0; e < E; ++e) {
        int d = (int)ei[E + e];
        deg[d] += 1.0f;
    }
    float dinv[N_NODES];
    for (int i = 0; i < N_NODES; ++i) dinv[i] = rsqrtf(deg[i]);
    for (int i = 0; i < N_NODES * N_NODES; ++i) P[i] = 0.0f;
    for (int e = 0; e < E; ++e) {
        int s = (int)ei[e];
        int d = (int)ei[E + e];
        P[d * N_NODES + s] += dinv[s] * dinv[d];
    }
    for (int n = 0; n < N_NODES; ++n)
        P[n * N_NODES + n] += dinv[n] * dinv[n];
}

// ---------------------------------------------------------------------------
// Prep 2: Wt[col][k] = f16(Wfc[k][col]) for k<19; k==19 carries bfc[col]
// (bias folded into the GEMM as an extra K row); else 0.  Layout gives each
// lane two aligned 16-byte B-fragment reads (K 8h..8h+7 and 16+8h..16+8h+7).
// ---------------------------------------------------------------------------
__global__ void build_Wt_kernel(const float* __restrict__ Wfc,
                                const float* __restrict__ bfc,
                                _Float16* __restrict__ Wt) {
    int i = blockIdx.x * blockDim.x + threadIdx.x;   // over FC_OUT*KPAD
    if (i >= FC_OUT * KPAD) return;
    int c = i >> 5;
    int k = i & 31;
    float v = (k < N_NODES) ? Wfc[k * FC_OUT + c]
            : (k == N_NODES) ? bfc[c] : 0.0f;
    Wt[(size_t)c * KPAD + k] = (_Float16)v;
}

// ---------------------------------------------------------------------------
// Phase 1: per-lane GCN (P^T broadcast from LDS, column-order accumulation),
// pack the WMMA A-fragment (16x32 f16, K19 = 1.0 bias row) and write 32B/lane
// to workspace.  No register cap: tiny total work, must not spill.
// ---------------------------------------------------------------------------
__global__ __launch_bounds__(256)
void gcn_afrag_kernel(const float* __restrict__ feat,
                      const float* __restrict__ W1, const float* __restrict__ b1,
                      const float* __restrict__ W2, const float* __restrict__ b2,
                      const float* __restrict__ P,
                      v16h* __restrict__ afrag, int ngroups) {
    __shared__ float sPT[N_NODES * N_NODES];   // sPT[s*19+n] = P[n*19+s]
    __shared__ float sW1[3], sb1[3], sW2[3], sb2;

    int tid = threadIdx.x;
    if (tid < N_NODES * N_NODES)               // transpose during the fill
        sPT[tid] = P[(tid % N_NODES) * N_NODES + (tid / N_NODES)];
    if (tid < 3) { sW1[tid] = W1[tid]; sb1[tid] = b1[tid]; sW2[tid] = W2[tid]; }
    if (tid == 3) sb2 = b2[0];
    __syncthreads();

    int wave = tid >> 5;
    int lane = tid & 31;
    int g = blockIdx.x * 8 + wave;          // 16-row group index (wave-uniform)
    if (g >= ngroups) return;               // whole wave exits together

    int mrow = lane & 15;                   // A-matrix M = lane % 16
    size_t row = (size_t)g * 16 + mrow;
    const float* fr = feat + row * N_NODES;

    float f[N_NODES], y[N_NODES];
#pragma unroll
    for (int n = 0; n < N_NODES; ++n)
        f[n] = __builtin_nontemporal_load(fr + n);   // read-once stream

    // conv1: y = P @ f, accumulated column-by-column (19 independent FMAs/step)
#pragma unroll
    for (int n = 0; n < N_NODES; ++n) y[n] = 0.0f;
#pragma unroll
    for (int s = 0; s < N_NODES; ++s) {
        float fs = f[s];
#pragma unroll
        for (int n = 0; n < N_NODES; ++n)
            y[n] = fmaf(sPT[s * N_NODES + n], fs, y[n]);
        asm volatile("" ::: "memory");      // confine this column's ds_loads
    }
    // elementwise: z[n] = sum_g tanh(y[n]*W1[g]+b1[g])*W2[g]  (reuse y[])
#pragma unroll
    for (int n = 0; n < N_NODES; ++n) {
        float acc = 0.0f;
#pragma unroll
        for (int gc = 0; gc < 3; ++gc) {
            float h1 = fast_tanh(fmaf(y[n], sW1[gc], sb1[gc]));
            acc = fmaf(h1, sW2[gc], acc);
        }
        y[n] = acc;                          // y now holds z
    }
    // conv2: h2 = tanh(P @ z + b2), same column-order accumulation (into f[])
#pragma unroll
    for (int n = 0; n < N_NODES; ++n) f[n] = sb2;
#pragma unroll
    for (int s = 0; s < N_NODES; ++s) {
        float zs = y[s];
#pragma unroll
        for (int n = 0; n < N_NODES; ++n)
            f[n] = fmaf(sPT[s * N_NODES + n], zs, f[n]);
        asm volatile("" ::: "memory");      // confine this column's ds_loads
    }
#pragma unroll
    for (int n = 0; n < N_NODES; ++n) f[n] = fast_tanh(f[n]);   // f now = h2

    // Pack A fragment (16-bit A 16x32 layout):
    //   lanes 0-15 : elems 0-7 = K0..7,  elems 8-15 = K16..23 (K19 = 1.0 bias)
    //   lanes 16-31: elems 0-7 = K8..15, elems 8-15 = K24..31 (all 0)
    v16h a;
    if (lane < 16) {
#pragma unroll
        for (int i = 0; i < 8; ++i) a[i] = (_Float16)f[i];
#pragma unroll
        for (int i = 0; i < 8; ++i) {
            int k = 16 + i;
            a[8 + i] = (k < N_NODES) ? (_Float16)f[k] : (_Float16)0.0f;
        }
        a[11] = (_Float16)1.0f;             // K=19: multiplies bfc row of Wt
    } else {
#pragma unroll
        for (int i = 0; i < 8; ++i) a[i] = (_Float16)f[8 + i];
#pragma unroll
        for (int i = 0; i < 8; ++i) a[8 + i] = (_Float16)0.0f;
    }

    afrag[(size_t)g * 32 + lane] = a;        // 32B/lane, 1KB/wave, L2-resident
}

// ---------------------------------------------------------------------------
// Phase 2: streaming WMMA FC.  ~40 VGPRs, no LDS, no cap needed -> max
// occupancy, pure store-bandwidth-bound (839 MB output @ 23.3 TB/s).
// ---------------------------------------------------------------------------
__global__ __launch_bounds__(256)
void wmma_fc_kernel(const v16h* __restrict__ afrag,
                    const _Float16* __restrict__ Wt,
                    float* __restrict__ out, int ngroups) {
    int wave = threadIdx.x >> 5;
    int lane = threadIdx.x & 31;
    int g = blockIdx.x * 8 + wave;
    if (g >= ngroups) return;               // whole wave exits together

    const v16h a = afrag[(size_t)g * 32 + lane];   // 2x global_load_b128

    int col0 = lane & 15;
    int half = lane >> 4;
    const _Float16* wtp = Wt + (size_t)col0 * KPAD + half * 8;
    float* outbase = out + (size_t)g * 16 * FC_OUT + (size_t)half * 8 * FC_OUT + col0;

#pragma unroll 1
    for (int t = 0; t < NTILES; ++t) {
        // B fragment: elems 0-7 = K{8h..8h+7}, elems 8-15 = K{16+8h..16+8h+7}
        typedef __attribute__((ext_vector_type(8))) _Float16 v8h;
        const v8h blo = *(const v8h*)(wtp + (size_t)t * 16 * KPAD);
        const v8h bhi = *(const v8h*)(wtp + (size_t)t * 16 * KPAD + 16);
        const v16h b = __builtin_shufflevector(blo, bhi, 0,1,2,3,4,5,6,7,8,9,10,11,12,13,14,15);

        v8f acc = {};                        // folds to inline SRC2=0 in v_wmma
        acc = __builtin_amdgcn_wmma_f32_16x16x32_f16(
            /*neg_a=*/false, a, /*neg_b=*/false, b,
            /*c_mod=*/(short)0, acc, /*reuse_a=*/false, /*reuse_b=*/false);

        float* o = outbase + t * 16;         // row M = v + 8*half
#pragma unroll
        for (int v = 0; v < 8; ++v)          // write-once stream: NT stores
            __builtin_nontemporal_store(acc[v], o + (size_t)v * FC_OUT);
    }
}

// ---------------------------------------------------------------------------
extern "C" void kernel_launch(void* const* d_in, const int* in_sizes, int n_in,
                              void* d_out, int out_size, void* d_ws, size_t ws_size,
                              hipStream_t stream) {
    const float*     feat = (const float*)d_in[0];
    const long long* ei   = (const long long*)d_in[1];
    const float*     W1   = (const float*)d_in[2];
    const float*     b1   = (const float*)d_in[3];
    const float*     W2   = (const float*)d_in[4];
    const float*     b2   = (const float*)d_in[5];
    const float*     Wfc  = (const float*)d_in[6];
    const float*     bfc  = (const float*)d_in[7];
    float*           out  = (float*)d_out;

    int E = in_sizes[1] / 2;                  // 128 edges (before self loops)
    int B = in_sizes[0] / N_NODES;            // 131072
    int ngroups = B / 16;                     // 8192 row-groups of 16

    float*    P     = (float*)d_ws;
    _Float16* Wt    = (_Float16*)((char*)d_ws + WT_OFFSET);
    v16h*     afrag = (v16h*)((char*)d_ws + AFRAG_OFFSET);   // 8 MB

    build_P_kernel<<<1, 32, 0, stream>>>(ei, E, P);

    int nWt = FC_OUT * KPAD;
    build_Wt_kernel<<<(nWt + 255) / 256, 256, 0, stream>>>(Wfc, bfc, Wt);

    int blocks = (ngroups + 7) / 8;           // 8 waves (16-row groups) per block
    gcn_afrag_kernel<<<blocks, 256, 0, stream>>>(feat, W1, b1, W2, b2,
                                                 P, afrag, ngroups);
    wmma_fc_kernel<<<blocks, 256, 0, stream>>>(afrag, Wt, out, ngroups);
}
